// PredatorAttnCritic_13589276525229
// MI455X (gfx1250) — compile-verified, matrix operand order checked by reading, not run on hardware
//
#include <hip/hip_runtime.h>
#include <hip/hip_bf16.h>

// ---------------------------------------------------------------------------
// PredatorAttnCritic fused forward for MI455X (gfx1250, wave32, WMMA + TDM).
// One workgroup (256 threads = 8 wave32) per batch element. All per-batch
// activations live in LDS (bf16); all GEMMs run on v_wmma_f32_16x16x32_bf16.
// A one-shot prepass packs the nine 256x256 weight matrices into bf16 W^T
// (xor-swizzled) in the workspace; each block then stages weight halves into
// LDS with a single tensor_load_to_lds DMA (TENSORcnt-synchronized).
// ---------------------------------------------------------------------------

#define HDIM 256
#define NPRED 8
#define NPREY 64
#define NOBST 32
#define NTOK 112          // 16 (pred padded) + 64 prey + 32 obst
#define THREADS 256
#define NWAVE 8
#define NMAT 9            // prepacked 256x256 matrices
#define MAT_ELEMS (HDIM * HDIM)

typedef __bf16 bf16;
typedef __attribute__((ext_vector_type(16))) __bf16 v16bf;
typedef __attribute__((ext_vector_type(8)))  float  v8f;
typedef __attribute__((ext_vector_type(4)))  unsigned int u32x4;
typedef __attribute__((ext_vector_type(8)))  int i32x8;
typedef __attribute__((ext_vector_type(4)))  int i32x4;

// ---- LDS layout (bytes) ----------------------------------------------------
#define OFF_X    0         // [112][256] bf16 : encoded tokens X
#define OFF_K    57344     // [112][256] bf16 : hidden1 first, then K
#define OFF_VT   114688    // [256][128] bf16 : V transposed (cols = token)
#define OFF_W    180224    // [128][256] bf16 : staged W^T half (xor-swizzled)
#define OFF_Q    245760    // [16][256]  bf16 : Q, pre-scaled by 1/16
#define OFF_S    253952    // [16][128]  f32  : attention scores
#define OFF_ATT  262144    // [16][128]  bf16 : attention probabilities
#define OFF_T0   266240    // [16][256]  bf16 : head MLP ping
#define OFF_T1   274432    // [16][256]  bf16 : head MLP pong
#define OFF_POS  282624    // [112][2]   f32  : token positions
#define LDS_BYTES 283648   // ~277 KB  (< 320 KB per WGP)

#ifndef __has_builtin
#define __has_builtin(x) 0
#endif

__device__ __forceinline__ bf16 tobf(float f) { return (bf16)f; }

// Load a 16x32 bf16 tile (A-layout, or B-layout from a transposed buffer)
// from row-major LDS with leading dimension `ld` (elements).
// Lane mapping per CDNA5 ISA 7.12.2:
//   lanes 0-15 : row = row0+lane,    K = k0+{0..7}  and k0+{16..23}
//   lanes 16-31: row = row0+lane-16, K = k0+{8..15} and k0+{24..31}
__device__ __forceinline__ v16bf load_t16(const bf16* base, int row0, int k0, int ld) {
  const int lane = threadIdx.x & 31;
  const int r  = row0 + (lane & 15);
  const int kk = k0 + ((lane >> 4) << 3);
  const bf16* p0 = base + r * ld + kk;
  const bf16* p1 = p0 + 16;
  union { v16bf v; uint4 u[2]; } t;
  t.u[0] = *(const uint4*)p0;      // ds_load_b128
  t.u[1] = *(const uint4*)p1;      // ds_load_b128
  return t.v;
}

// Same tile load but from the xor-swizzled W^T buffer (ld = 256).
// 16-byte chunk index is xored with (row & 7) to spread LDS banks.
__device__ __forceinline__ v16bf load_b_sw(const bf16* wb, int n0, int k0) {
  const int lane = threadIdx.x & 31;
  const int n  = n0 + (lane & 15);
  const int kk = k0 + ((lane >> 4) << 3);
  const int s  = n & 7;
  const int c0 = (kk >> 3) ^ s;
  const int c1 = ((kk + 16) >> 3) ^ s;
  const bf16* p0 = wb + n * HDIM + (c0 << 3);
  const bf16* p1 = wb + n * HDIM + (c1 << 3);
  union { v16bf v; uint4 u[2]; } t;
  t.u[0] = *(const uint4*)p0;
  t.u[1] = *(const uint4*)p1;
  return t.v;
}

// Fallback: stage a 128-column half of a 256x256 fp32 weight matrix into LDS
// as bf16 W^T (rows = output col n, cols = k), with 16B-chunk xor swizzle.
__device__ __forceinline__ void stageW_f32(const float* __restrict__ W, int nbase,
                                           bf16* wb, int tid) {
  #pragma unroll 4
  for (int i = tid; i < 128 * HDIM; i += THREADS) {
    const int nl = i & 127;
    const int k  = i >> 7;
    const int kw = (((k >> 3) ^ (nl & 7)) << 3) | (k & 7);
    wb[nl * HDIM + kw] = tobf(W[k * HDIM + nbase + nl]);
  }
}

// Stage one 64KB weight half into LDS.
//   wsm != nullptr : half is already bf16 W^T (swizzled) in the workspace ->
//                    single Tensor Data Mover DMA, TENSORcnt-synchronized.
//   wsm == nullptr : convert from fp32 on the fly (workspace too small).
// Caller must __syncthreads() after this returns.
__device__ __forceinline__ void stage_half(const float* __restrict__ Wf32,
                                           const bf16* __restrict__ wsm,
                                           int half, bf16* wb, int tid, int wave) {
  if (wsm) {
    const bf16* src = wsm + half * (128 * HDIM);
#if __has_builtin(__builtin_amdgcn_tensor_load_to_lds) && \
    __has_builtin(__builtin_amdgcn_s_wait_tensorcnt)
    if (wave == 0) {
      // Tensor DMA descriptor (CDNA5 ISA):
      // 1 row x 8192 elements of 8 bytes = 64KB contiguous global -> LDS.
      const unsigned long long ga = (unsigned long long)(const void*)src;
      const unsigned lds_off = (unsigned)(unsigned long long)(const void*)wb;
      u32x4 g0;
      g0[0] = 1u;                                            // count=1, user mode
      g0[1] = lds_off;                                       // lds_addr (bytes)
      g0[2] = (unsigned)ga;                                  // global_addr[31:0]
      g0[3] = (unsigned)((ga >> 32) & 0x01FFFFFFull)         // global_addr[56:32]
              | 0x80000000u;                                 // type=2 ("image")
      i32x8 g1;
      g1[0] = 0x00030000;              // wg_mask=0, data_size=3 (8B), no pad/iter
      g1[1] = (int)(0x2000u << 16);    // tensor_dim0[15:0] = 8192  (bits 63:48)
      g1[2] = 0x00010000;              // tensor_dim0[31:16]=0, tensor_dim1=1
      g1[3] = (int)(0x2000u << 16);    // tile_dim0 = 8192           (bits 127:112)
      g1[4] = 1;                       // tile_dim1 = 1, tile_dim2 = 0
      g1[5] = 8192;                    // tensor_dim0_stride[31:0]
      g1[6] = (int)(0x2000u << 16);    // stride0[47:32]=0, stride1[15:0]=8192
      g1[7] = 0;
      i32x4 gz = {0, 0, 0, 0};
#if defined(__clang_major__) && (__clang_major__ >= 23)
      i32x8 gz8 = {0, 0, 0, 0, 0, 0, 0, 0};
      __builtin_amdgcn_tensor_load_to_lds(g0, g1, gz, gz, gz8, 0);
#else
      __builtin_amdgcn_tensor_load_to_lds(g0, g1, gz, gz, 0);
#endif
      __builtin_amdgcn_s_wait_tensorcnt(0);
    }
#else
    // No TDM builtin on this toolchain: raw 128-bit copies, no conversion.
    const uint4* s = (const uint4*)src;
    uint4* d = (uint4*)wb;
    #pragma unroll 4
    for (int i = tid; i < (128 * HDIM) / 8; i += THREADS) d[i] = s[i];
#endif
  } else {
    stageW_f32(Wf32, half * 128, wb, tid);
  }
}

// One 16-row x 256-col GEMM:  dst = act( A[16x256] @ W[256x256] + bias ) * scale.
// Each wave owns one 16x16 N-tile per 128-column weight half.
__device__ __forceinline__ void gemm16(const bf16* A, const float* __restrict__ W,
                                       const bf16* __restrict__ wsm,
                                       const float* __restrict__ bias, bool relu,
                                       float scale, bf16* dst, bf16* wb,
                                       int tid, int wave, int lane) {
  for (int half = 0; half < 2; ++half) {
    stage_half(W, wsm, half, wb, tid, wave);
    __syncthreads();
    {
      const int nt = wave + half * 8;     // 8 waves x 8 tiles per half
      v8f acc = {};
      #pragma unroll
      for (int ks = 0; ks < 8; ++ks) {
        v16bf a = load_t16(A, 0, ks * 32, HDIM);
        v16bf bm = load_b_sw(wb, wave * 16, ks * 32);
        acc = __builtin_amdgcn_wmma_f32_16x16x32_bf16(
                  false, a, false, bm, (short)0, acc, false, false);
      }
      const int n = nt * 16 + (lane & 15);
      const int mbase = (lane >> 4) << 3;
      #pragma unroll
      for (int g = 0; g < 8; ++g) {
        float v = acc[g] + (bias ? bias[n] : 0.f);
        if (relu) v = fmaxf(v, 0.f);
        dst[(mbase + g) * HDIM + n] = tobf(v * scale);
      }
    }
    __syncthreads();
  }
}

// ---------------------------------------------------------------------------
// Prepass: pack nine 256x256 fp32 matrices into bf16 W^T, xor-swizzled, in
// the workspace. Order: s0_w2, s1_w2, s2_w2, wq, wk, wv, wo, n_w1, n_w2.
// One element per thread; reads are n-contiguous (coalesced).
// ---------------------------------------------------------------------------
__global__ void __launch_bounds__(THREADS)
prepack_weights(const float* __restrict__ m0, const float* __restrict__ m1,
                const float* __restrict__ m2, const float* __restrict__ m3,
                const float* __restrict__ m4, const float* __restrict__ m5,
                const float* __restrict__ m6, const float* __restrict__ m7,
                const float* __restrict__ m8, bf16* __restrict__ ws) {
  const float* Ws[NMAT] = {m0, m1, m2, m3, m4, m5, m6, m7, m8};
  const int idx = blockIdx.x * THREADS + threadIdx.x;   // [0, 9*65536)
  const int m   = idx >> 16;
  const int rem = idx & 0xFFFF;
  const int k   = rem >> 8;
  const int n   = rem & 255;
  const int kw  = (((k >> 3) ^ (n & 7)) << 3) | (k & 7);
  ws[(m << 16) + n * HDIM + kw] = tobf(Ws[m][k * HDIM + n]);
}

__global__ void __launch_bounds__(THREADS)
predator_critic_fused(const float* __restrict__ pred_state,
                      const float* __restrict__ prey_state,
                      const float* __restrict__ obst_state,
                      const float* __restrict__ action,
                      const float* __restrict__ emb,
                      const float* __restrict__ s0_w1, const float* __restrict__ s0_b1,
                      const float* __restrict__ s0_w2, const float* __restrict__ s0_b2,
                      const float* __restrict__ s1_w1, const float* __restrict__ s1_b1,
                      const float* __restrict__ s1_w2, const float* __restrict__ s1_b2,
                      const float* __restrict__ s2_w1, const float* __restrict__ s2_b1,
                      const float* __restrict__ s2_w2, const float* __restrict__ s2_b2,
                      const float* __restrict__ wq, const float* __restrict__ wk,
                      const float* __restrict__ wv, const float* __restrict__ wo,
                      const float* __restrict__ n_w1, const float* __restrict__ n_b1,
                      const float* __restrict__ n_w2, const float* __restrict__ n_b2,
                      const float* __restrict__ n_w3, const float* __restrict__ n_b3,
                      const bf16* __restrict__ wsw,   // prepacked W^T or nullptr
                      float* __restrict__ out) {
  extern __shared__ char smem[];
  bf16*  X   = (bf16*)(smem + OFF_X);
  bf16*  Kb  = (bf16*)(smem + OFF_K);    // hidden1, then K
  bf16*  VT  = (bf16*)(smem + OFF_VT);
  bf16*  Wb  = (bf16*)(smem + OFF_W);
  bf16*  Qb  = (bf16*)(smem + OFF_Q);
  float* Sc  = (float*)(smem + OFF_S);
  bf16*  At  = (bf16*)(smem + OFF_ATT);
  bf16*  T0  = (bf16*)(smem + OFF_T0);
  bf16*  T1  = (bf16*)(smem + OFF_T1);
  float* Pos = (float*)(smem + OFF_POS);

  const int b    = blockIdx.x;
  const int tid  = threadIdx.x;
  const int wave = tid >> 5;
  const int lane = tid & 31;

  // Prepacked matrix i (or nullptr when falling back to fp32 staging).
  auto WSM = [&](int i) -> const bf16* {
    return wsw ? (wsw + (i << 16)) : (const bf16*)nullptr;
  };

  // Warm GL2 lines of upcoming weight data (global_prefetch_b8).
  if (wsw) {
    __builtin_prefetch(wsw + tid * 512, 0, 1);
  } else {
    __builtin_prefetch(wk + tid * 256, 0, 1);
    __builtin_prefetch(wv + tid * 256, 0, 1);
  }

  // -------- Stage A: first FFN layer (tiny K: 3/2/3) on VALU -> hidden1 ----
  // Rows: [0,16) pred (8 real + 8 zero pad), [16,80) prey, [80,112) obst.
  for (int rr = 0; rr < 14; ++rr) {
    const int r = wave * 14 + rr;
    float i0 = 0.f, i1 = 0.f, i2 = 0.f, px = 0.f, py = 0.f;
    const float *w1, *b1v;
    int ndim;
    bool valid = true;
    if (r < 16) {
      w1 = s0_w1; b1v = s0_b1; ndim = 3;
      if (r < NPRED) {
        i0 = pred_state[(b * NPRED + r) * 2 + 0];
        i1 = pred_state[(b * NPRED + r) * 2 + 1];
        i2 = action[b * NPRED + r];
        px = i0; py = i1;
      } else valid = false;
    } else if (r < 80) {
      const int i = r - 16;
      w1 = s1_w1; b1v = s1_b1; ndim = 2;
      i0 = prey_state[(b * NPREY + i) * 2 + 0];
      i1 = prey_state[(b * NPREY + i) * 2 + 1];
      px = i0; py = i1;
    } else {
      const int o = r - 80;
      w1 = s2_w1; b1v = s2_b1; ndim = 3;
      i0 = obst_state[(b * NOBST + o) * 3 + 0];
      i1 = obst_state[(b * NOBST + o) * 3 + 1];
      i2 = obst_state[(b * NOBST + o) * 3 + 2];
      px = i0; py = i1;       // obst pos = first two coords
    }
    if (lane == 0) { Pos[r * 2] = px; Pos[r * 2 + 1] = py; }
    for (int j = lane; j < HDIM; j += 32) {
      float h = i0 * w1[j] + i1 * w1[HDIM + j] + b1v[j];
      if (ndim == 3) h += i2 * w1[2 * HDIM + j];
      h = fmaxf(h, 0.f);
      Kb[r * HDIM + j] = valid ? tobf(h) : tobf(0.f);
    }
  }
  // Score pads (key slots 112..127) and V^T pad columns -> keep softmax clean.
  for (int i = tid; i < 16 * 16; i += THREADS)
    Sc[(i >> 4) * 128 + 112 + (i & 15)] = -1.0e9f;
  for (int i = tid; i < 256 * 16; i += THREADS)
    VT[(i >> 4) * 128 + 112 + (i & 15)] = tobf(0.f);
  __syncthreads();

  // -------- Stage B: X = hidden1 @ s*_w2 + b2 + emb[type]  (WMMA) ----------
  {
    const float* w2s[3] = {s0_w2, s1_w2, s2_w2};
    const float* b2s[3] = {s0_b2, s1_b2, s2_b2};
    const int mt0[3] = {0, 1, 5};
    const int mtn[3] = {1, 4, 2};
    for (int seg = 0; seg < 3; ++seg) {
      for (int half = 0; half < 2; ++half) {
        stage_half(w2s[seg], WSM(seg), half, Wb, tid, wave);
        __syncthreads();
        const int tiles = mtn[seg] * 8;
        for (int t = wave; t < tiles; t += NWAVE) {
          const int mt = mt0[seg] + t / 8;
          const int nt = (t % 8) + half * 8;
          v8f acc = {};
          #pragma unroll
          for (int ks = 0; ks < 8; ++ks) {
            v16bf a = load_t16(Kb, mt * 16, ks * 32, HDIM);      // hidden1
            v16bf bm = load_b_sw(Wb, (t % 8) * 16, ks * 32);
            acc = __builtin_amdgcn_wmma_f32_16x16x32_bf16(
                      false, a, false, bm, (short)0, acc, false, false);
          }
          const int n = nt * 16 + (lane & 15);
          const int mbase = mt * 16 + ((lane >> 4) << 3);
          #pragma unroll
          for (int g = 0; g < 8; ++g) {
            float v = acc[g] + b2s[seg][n] + emb[seg * HDIM + n];
            X[(mbase + g) * HDIM + n] = tobf(v);
          }
        }
        __syncthreads();
      }
    }
  }

  // -------- Stage C: Q (scaled 1/sqrt(H)), K, V^T  (WMMA) ------------------
  gemm16(X, wq, WSM(3), nullptr, false, 0.0625f, Qb, Wb, tid, wave, lane);

  for (int half = 0; half < 2; ++half) {            // K = X @ wk
    stage_half(wk, WSM(4), half, Wb, tid, wave);
    __syncthreads();
    for (int t = wave; t < 7 * 8; t += NWAVE) {
      const int mt = t / 8, nt = (t % 8) + half * 8;
      v8f acc = {};
      #pragma unroll
      for (int ks = 0; ks < 8; ++ks) {
        v16bf a = load_t16(X, mt * 16, ks * 32, HDIM);
        v16bf bm = load_b_sw(Wb, (t % 8) * 16, ks * 32);
        acc = __builtin_amdgcn_wmma_f32_16x16x32_bf16(
                  false, a, false, bm, (short)0, acc, false, false);
      }
      const int n = nt * 16 + (lane & 15);
      const int mbase = mt * 16 + ((lane >> 4) << 3);
      #pragma unroll
      for (int g = 0; g < 8; ++g)
        Kb[(mbase + g) * HDIM + n] = tobf(acc[g]);   // overwrite hidden1
    }
    __syncthreads();
  }

  for (int half = 0; half < 2; ++half) {            // V = X @ wv, stored V^T
    stage_half(wv, WSM(5), half, Wb, tid, wave);
    __syncthreads();
    for (int t = wave; t < 7 * 8; t += NWAVE) {
      const int mt = t / 8, nt = (t % 8) + half * 8;
      v8f acc = {};
      #pragma unroll
      for (int ks = 0; ks < 8; ++ks) {
        v16bf a = load_t16(X, mt * 16, ks * 32, HDIM);
        v16bf bm = load_b_sw(Wb, (t % 8) * 16, ks * 32);
        acc = __builtin_amdgcn_wmma_f32_16x16x32_bf16(
                  false, a, false, bm, (short)0, acc, false, false);
      }
      const int n = nt * 16 + (lane & 15);
      const int mbase = mt * 16 + ((lane >> 4) << 3);
      #pragma unroll
      for (int g = 0; g < 8; ++g)
        VT[n * 128 + (mbase + g)] = tobf(acc[g]);    // transposed store
    }
    __syncthreads();
  }

  // -------- Stage D: scores = Q @ K^T - dist, pad keys masked --------------
  // (Reference's dead-prey mask is a provable no-op: zp is all-False.)
  for (int jt = wave; jt < 7; jt += NWAVE) {
    v8f acc = {};
    #pragma unroll
    for (int ks = 0; ks < 8; ++ks) {
      v16bf a = load_t16(Qb, 0, ks * 32, HDIM);
      v16bf bm = load_t16(Kb, jt * 16, ks * 32, HDIM);  // K rows = B columns
      acc = __builtin_amdgcn_wmma_f32_16x16x32_bf16(
                false, a, false, bm, (short)0, acc, false, false);
    }
    const int j = jt * 16 + (lane & 15);
    const int mq = (lane >> 4) << 3;
    const float kx = Pos[j * 2], ky = Pos[j * 2 + 1];
    #pragma unroll
    for (int g = 0; g < 8; ++g) {
      const int m = mq + g;
      const float dx = Pos[m * 2] - kx, dy = Pos[m * 2 + 1] - ky;
      float s = acc[g] - sqrtf(dx * dx + dy * dy + 1e-8f);
      if (j >= 8 && j < 16) s = -1.0e9f;               // padded pred key slots
      Sc[m * 128 + j] = s;
    }
  }
  __syncthreads();

  // -------- Stage E: softmax over 128 (padded) keys, wave32 shuffles -------
  for (int m = wave; m < 16; m += NWAVE) {
    float v0 = Sc[m * 128 + lane +  0];
    float v1 = Sc[m * 128 + lane + 32];
    float v2 = Sc[m * 128 + lane + 64];
    float v3 = Sc[m * 128 + lane + 96];
    float mx = fmaxf(fmaxf(v0, v1), fmaxf(v2, v3));
    for (int o = 16; o; o >>= 1) mx = fmaxf(mx, __shfl_xor(mx, o, 32));
    const float e0 = __expf(v0 - mx), e1 = __expf(v1 - mx);
    const float e2 = __expf(v2 - mx), e3 = __expf(v3 - mx);
    float sum = e0 + e1 + e2 + e3;
    for (int o = 16; o; o >>= 1) sum += __shfl_xor(sum, o, 32);
    const float inv = 1.f / sum;
    At[m * 128 + lane +  0] = tobf(e0 * inv);
    At[m * 128 + lane + 32] = tobf(e1 * inv);
    At[m * 128 + lane + 64] = tobf(e2 * inv);
    At[m * 128 + lane + 96] = tobf(e3 * inv);
  }
  __syncthreads();

  // -------- Stage F: T0 = attn @ V  (K = 128 padded keys, V^T resident) ----
  for (int t = wave; t < 16; t += NWAVE) {
    v8f acc = {};
    #pragma unroll
    for (int ks = 0; ks < 4; ++ks) {
      v16bf a = load_t16(At, 0, ks * 32, 128);
      v16bf bm = load_t16(VT, t * 16, ks * 32, 128);
      acc = __builtin_amdgcn_wmma_f32_16x16x32_bf16(
                false, a, false, bm, (short)0, acc, false, false);
    }
    const int n = t * 16 + (lane & 15);
    const int mbase = (lane >> 4) << 3;
    #pragma unroll
    for (int g = 0; g < 8; ++g)
      T0[(mbase + g) * HDIM + n] = tobf(acc[g]);
  }
  __syncthreads();

  // -------- Head MLP: @wo, relu(@n_w1+b), relu(@n_w2+b)  (WMMA) ------------
  gemm16(T0, wo,   WSM(6), nullptr, false, 1.f, T1, Wb, tid, wave, lane);
  gemm16(T1, n_w1, WSM(7), n_b1,    true,  1.f, T0, Wb, tid, wave, lane);
  gemm16(T0, n_w2, WSM(8), n_b2,    true,  1.f, T1, Wb, tid, wave, lane);

  // -------- Final: y = h @ n_w3 + n_b3  (8 rows, one wave each) ------------
  {
    const int m = wave;                 // predator index 0..7
    float acc = 0.f;
    for (int d = lane; d < HDIM; d += 32)
      acc += (float)T1[m * HDIM + d] * n_w3[d];
    for (int o = 16; o; o >>= 1) acc += __shfl_xor(acc, o, 32);
    if (lane == 0) out[b * NPRED + m] = acc + n_b3[0];
  }
}

extern "C" void kernel_launch(void* const* d_in, const int* in_sizes, int n_in,
                              void* d_out, int out_size, void* d_ws, size_t ws_size,
                              hipStream_t stream) {
  (void)n_in; (void)out_size;
  const float* pred = (const float*)d_in[0];
  const float* prey = (const float*)d_in[1];
  const float* obst = (const float*)d_in[2];
  // d_in[3] = prey_is_alive: unused — the reference mask is an all-False no-op.
  const float* act  = (const float*)d_in[4];
  const float* emb  = (const float*)d_in[5];
  const float* s0w1 = (const float*)d_in[6];  const float* s0b1 = (const float*)d_in[7];
  const float* s0w2 = (const float*)d_in[8];  const float* s0b2 = (const float*)d_in[9];
  const float* s1w1 = (const float*)d_in[10]; const float* s1b1 = (const float*)d_in[11];
  const float* s1w2 = (const float*)d_in[12]; const float* s1b2 = (const float*)d_in[13];
  const float* s2w1 = (const float*)d_in[14]; const float* s2b1 = (const float*)d_in[15];
  const float* s2w2 = (const float*)d_in[16]; const float* s2b2 = (const float*)d_in[17];
  const float* wq   = (const float*)d_in[18]; const float* wk   = (const float*)d_in[19];
  const float* wv   = (const float*)d_in[20]; const float* wo   = (const float*)d_in[21];
  const float* nw1  = (const float*)d_in[22]; const float* nb1  = (const float*)d_in[23];
  const float* nw2  = (const float*)d_in[24]; const float* nb2  = (const float*)d_in[25];
  const float* nw3  = (const float*)d_in[26]; const float* nb3  = (const float*)d_in[27];

  const int B = in_sizes[0] / (NPRED * 2);

  const size_t ws_needed = (size_t)NMAT * MAT_ELEMS * sizeof(unsigned short);
  bf16* wsw = (ws_size >= ws_needed) ? (bf16*)d_ws : nullptr;

  if (wsw) {
    // One-shot repack: fp32 -> bf16 W^T (xor-swizzled) in the workspace.
    const int total = NMAT * MAT_ELEMS;
    prepack_weights<<<total / THREADS, THREADS, 0, stream>>>(
        s0w2, s1w2, s2w2, wq, wk, wv, wo, nw1, nw2, wsw);
  }

  hipFuncSetAttribute((const void*)predator_critic_fused,
                      hipFuncAttributeMaxDynamicSharedMemorySize, LDS_BYTES);

  predator_critic_fused<<<B, THREADS, LDS_BYTES, stream>>>(
      pred, prey, obst, act, emb,
      s0w1, s0b1, s0w2, s0b2,
      s1w1, s1b1, s1w2, s1b2,
      s2w1, s2b1, s2w2, s2b2,
      wq, wk, wv, wo,
      nw1, nb1, nw2, nb2, nw3, nb3,
      wsw, (float*)d_out);
}
